// MultiHeadAttention_7541962572058
// MI455X (gfx1250) — matrix operand order repaired and checked by reading, not compiled, not run on hardware
//
#include <hip/hip_runtime.h>
#include <math.h>
#include <stdint.h>

// ---------------------------------------------------------------------------
// MI455X (gfx1250) implementation. wave32. Heavy GEMMs run on
// v_wmma_f32_16x16x32_bf16 (f32 accumulate). LDS tiles are staged with CDNA5
// async global->LDS loads (ASYNCcnt), double-buffered so the next K-tile's
// transfers overlap the current tile's WMMAs.
// ---------------------------------------------------------------------------

#define LN2F 0.6931471805599453f

typedef __attribute__((ext_vector_type(16))) __bf16 v16bf;
typedef __attribute__((ext_vector_type(8)))  float  v8f;

struct alignas(16) U4 { unsigned int a, b, c, d; };
struct alignas(16) F4 { float a, b, c, d; };

union Frag {
  v16bf        v;
  unsigned int u[8];
  U4           q[2];
};

__device__ __forceinline__ unsigned short f2bf(float f) {
  union { float f; unsigned int u; } cv; cv.f = f;
  return (unsigned short)((cv.u + 0x7FFFu + ((cv.u >> 16) & 1u)) >> 16);
}
__device__ __forceinline__ float bf2f(unsigned short h) {
  union { unsigned int u; float f; } cv; cv.u = ((unsigned int)h) << 16;
  return cv.f;
}
__device__ __forceinline__ unsigned int pack2(float lo, float hi) {
  return (unsigned int)f2bf(lo) | ((unsigned int)f2bf(hi) << 16);
}
__device__ __forceinline__ float ssp_m_ln2(float x) {   // softplus(x) - ln2
  float t = (x > 20.0f) ? x : log1pf(__expf(x));
  return t - LN2F;
}
__device__ __forceinline__ v8f wmma_bf16(const Frag& A, const Frag& B, v8f C) {
  return __builtin_amdgcn_wmma_f32_16x16x32_bf16(false, A.v, false, B.v,
                                                 (short)0, C, false, false);
}

// CDNA5 async global->LDS 16B copy (tracked by ASYNCcnt; no VGPR data path).
// The low 32 bits of a generic LDS pointer are the LDS byte address.
__device__ __forceinline__ void async_b128(void* lds, const void* g) {
  unsigned loff = (unsigned)(uintptr_t)lds;
  unsigned long long ga = (unsigned long long)(uintptr_t)g;
  asm volatile("global_load_async_to_lds_b128 %0, %1, off"
               :: "v"(loff), "v"(ga) : "memory");
}

// ---------------------------------------------------------------------------
// fp32 [K,N] -> bf16 transposed [N,K] (weights; done once, tiny)
// ---------------------------------------------------------------------------
__global__ void cvtT_f32_bf16(const float* __restrict__ s,
                              unsigned short* __restrict__ d, int K, int N) {
  size_t tot = (size_t)K * N;
  size_t i  = (size_t)blockIdx.x * blockDim.x + threadIdx.x;
  size_t st = (size_t)gridDim.x * blockDim.x;
  for (; i < tot; i += st) {
    size_t k = i / (size_t)N, n = i % (size_t)N;
    d[n * (size_t)K + k] = f2bf(s[i]);
  }
}

// ---------------------------------------------------------------------------
// Generic WMMA GEMM:  C[M,Nn] = act( A[M,K] @ W[K,Nn] + bias )
//  - Weights pre-transposed: WT[Nn,K] bf16 -> B tile staging is a contiguous
//    row-major copy, done with global_load_async_to_lds_b128.
//  - A source: fp32 (converted synchronously while staging), bf16 (async), or
//    gather-concat mode (row e = [outbf[b,i,:] , outbf[b,j,:]], async).
//  - 256 threads = 8 waves; block tile 128x64; wave tile 32x32 (2x2 WMMA);
//    K staged 64 at a time -> 8 WMMA per barrier round.
//  - Double-buffered LDS: tile t+1's async loads are issued before waiting
//    for tile t (ASYNCcnt completes in order), hiding global latency.
// ---------------------------------------------------------------------------
__global__ __launch_bounds__(256)
void gemm_wmma_bf16(const float* __restrict__ Af32,
                    const unsigned short* __restrict__ Ab16,
                    const unsigned short* __restrict__ gatherSrc,
                    const int* __restrict__ pairs, int En,
                    const unsigned short* __restrict__ WT,   // [Nn][K]
                    const float* __restrict__ bias,
                    float* __restrict__ outF,
                    unsigned short* __restrict__ outB,
                    int M, int Nn, int K, int act) {
  __shared__ alignas(16) unsigned short As[2][128][80];  // 64 cols used + pad
  __shared__ alignas(16) unsigned short Bs[2][64][80];   // [n][k]

  const int tid  = threadIdx.x;
  const int lane = tid & 31;
  const int wave = tid >> 5;
  const int wm   = wave & 3;            // 4 wave-rows of 32
  const int wn   = wave >> 2;           // 2 wave-cols of 32
  const int rowBase = blockIdx.y * 128;
  const int colBase = blockIdx.x * 64;
  const int m16  = lane & 15;
  const int half = lane >> 4;

  v8f acc[2][2] = {};

  // Stage K-tile t into LDS buffer t&1. Async paths issue 6 ASYNCcnt events
  // per wave per tile (4 A chunks + 2 B chunks); fp32-A path issues 2 (B only).
  auto stage = [&](int t) {
    const int ktl = t * 64;
    const int pp  = t & 1;
    if (Af32) {                               // convert on the fly (sync)
      for (int c = tid; c < 1024; c += 256) {
        int r = c >> 3, cc = (c & 7) << 3;
        const float* ap = Af32 + (size_t)(rowBase + r) * K + ktl + cc;
        F4 f0 = *(const F4*)ap;
        F4 f1 = *(const F4*)(ap + 4);
        U4 val;
        val.a = pack2(f0.a, f0.b); val.b = pack2(f0.c, f0.d);
        val.c = pack2(f1.a, f1.b); val.d = pack2(f1.c, f1.d);
        *(U4*)&As[pp][r][cc] = val;
      }
    } else if (gatherSrc) {                   // nf concat-gather (async)
      for (int c = tid; c < 1024; c += 256) {
        int r = c >> 3, cc = (c & 7) << 3;
        int grow = rowBase + r, gcol = ktl + cc;
        int bi   = pairs[grow];
        int node = (gcol < 512) ? pairs[En + grow] : pairs[2 * En + grow];
        const unsigned short* src =
            gatherSrc + ((size_t)(bi * 512 + node)) * 512 + (gcol & 511);
        async_b128(&As[pp][r][cc], src);
      }
    } else {                                  // plain bf16 A (async)
      for (int c = tid; c < 1024; c += 256) {
        int r = c >> 3, cc = (c & 7) << 3;
        async_b128(&As[pp][r][cc], Ab16 + (size_t)(rowBase + r) * K + ktl + cc);
      }
    }
    for (int c = tid; c < 512; c += 256) {    // B tile from WT rows (async)
      int n = c >> 3, cc = (c & 7) << 3;
      async_b128(&Bs[pp][n][cc], WT + (size_t)(colBase + n) * K + ktl + cc);
    }
  };

  const int T = K >> 6;                       // K / 64 tiles
  stage(0);

  for (int t = 0; t < T; ++t) {
    const bool more = (t + 1 < T);
    if (more) stage(t + 1);                   // issue next tile's transfers

    // Wait only for tile t's async transfers (in-order completion), leaving
    // tile t+1's in flight behind the WMMAs below.
    if (more) {
      if (Af32) asm volatile("s_wait_asynccnt 0x2" ::: "memory");
      else      asm volatile("s_wait_asynccnt 0x6" ::: "memory");
    } else {
      asm volatile("s_wait_asynccnt 0x0" ::: "memory");
    }
    __syncthreads();

    const int pp = t & 1;
    // ---- 2 K-substeps of 32: fragments are contiguous 16B LDS loads ----
#pragma unroll
    for (int ks = 0; ks < 2; ++ks) {
      const int ko = ks * 32;
      Frag a0, a1, b0, b1;
      const unsigned short* ap0 = &As[pp][wm * 32 + m16][ko + half * 16];
      a0.q[0] = *(const U4*)ap0;
      a0.q[1] = *(const U4*)(ap0 + 8);
      const unsigned short* ap1 = &As[pp][wm * 32 + 16 + m16][ko + half * 16];
      a1.q[0] = *(const U4*)ap1;
      a1.q[1] = *(const U4*)(ap1 + 8);
      const unsigned short* bp0 = &Bs[pp][wn * 32 + m16][ko + half * 16];
      b0.q[0] = *(const U4*)bp0;
      b0.q[1] = *(const U4*)(bp0 + 8);
      const unsigned short* bp1 = &Bs[pp][wn * 32 + 16 + m16][ko + half * 16];
      b1.q[0] = *(const U4*)bp1;
      b1.q[1] = *(const U4*)(bp1 + 8);

      acc[0][0] = wmma_bf16(a0, b0, acc[0][0]);
      acc[0][1] = wmma_bf16(a0, b1, acc[0][1]);
      acc[1][0] = wmma_bf16(a1, b0, acc[1][0]);
      acc[1][1] = wmma_bf16(a1, b1, acc[1][1]);
    }
    __syncthreads();   // all waves done reading buffer pp before it is reused
  }

  // ---- epilogue: bias + optional softplus-ln2, f32 and/or bf16 stores ----
  for (int i = 0; i < 2; ++i) {
    for (int j = 0; j < 2; ++j) {
      int col  = colBase + wn * 32 + j * 16 + m16;
      float bv = bias ? bias[col] : 0.0f;
      int row0 = rowBase + wm * 32 + i * 16 + half * 8;
#pragma unroll
      for (int r = 0; r < 8; ++r) {
        float x = acc[i][j][r] + bv;
        if (act) x = ssp_m_ln2(x);
        size_t idx = (size_t)(row0 + r) * Nn + col;
        if (outF) outF[idx] = x;
        if (outB) outB[idx] = f2bf(x);
      }
    }
  }
}

// ---------------------------------------------------------------------------
// Global attention (heads 0..3). One wave per (b, h, 16-row q tile).
//  S = (Q/8) K^T via WMMA; pre-mask head-0 scores -> top_score; softmax in
//  LDS; P (bf16) @ V via WMMA -> ctx cols [h*64, h*64+64).
// ---------------------------------------------------------------------------
__global__ __launch_bounds__(32)
void global_attn(const unsigned short* __restrict__ qbf,
                 const unsigned short* __restrict__ kbf,
                 const unsigned short* __restrict__ vbf,
                 const unsigned char* __restrict__ maskp,
                 unsigned short* __restrict__ ctxbf,
                 float* __restrict__ topsc) {
  __shared__ alignas(16) float          sc[16][512];
  __shared__ alignas(16) unsigned short pb[16][512];

  const int lane = threadIdx.x;
  const int m16  = lane & 15;
  const int half = lane >> 4;
  const int rb   = blockIdx.x * 16;   // q-row tile
  const int h    = blockIdx.y;        // head 0..3
  const int bb   = blockIdx.z;        // batch

  Frag aq[2];
  {
    const unsigned short* qrow =
        qbf + ((size_t)(bb * 512 + rb + m16)) * 512 + h * 64;
#pragma unroll
    for (int ks = 0; ks < 2; ++ks) {
      const unsigned short* qp = qrow + ks * 32;
      aq[ks].q[0] = *(const U4*)(qp + half * 16);
      aq[ks].q[1] = *(const U4*)(qp + half * 16 + 8);
    }
  }

  // ---- scores: S[16,512] ----
  for (int jt = 0; jt < 512; jt += 16) {
    v8f s = {};
#pragma unroll
    for (int ks = 0; ks < 2; ++ks) {
      Frag bk;   // B 32x16: col = key index, K = feature dim (contiguous)
      const unsigned short* kp =
          kbf + ((size_t)(bb * 512 + jt + m16)) * 512 + h * 64 + ks * 32 + half * 16;
      bk.q[0] = *(const U4*)kp;
      bk.q[1] = *(const U4*)(kp + 8);
      s = wmma_bf16(aq[ks], bk, s);
    }
    int col = jt + m16;
#pragma unroll
    for (int r = 0; r < 8; ++r) {
      int rr = half * 8 + r;
      float v = s[r] * 0.125f;                     // 1/sqrt(DH)
      size_t gidx = ((size_t)(bb * 512 + rb + rr)) * 512 + col;
      if (h == 0) topsc[gidx] = v;                 // pre-mask, head 0
      if (maskp[gidx]) v = -1e18f;
      sc[rr][col] = v;
    }
  }
  __syncthreads();

  // ---- row softmax, probabilities stored bf16 in LDS ----
  for (int r = 0; r < 16; ++r) {
    float mx = -1e30f;
#pragma unroll
    for (int ci = 0; ci < 16; ++ci) mx = fmaxf(mx, sc[r][lane + ci * 32]);
    for (int o = 16; o >= 1; o >>= 1) mx = fmaxf(mx, __shfl_xor(mx, o, 32));
    float ev[16];
    float sum = 0.0f;
#pragma unroll
    for (int ci = 0; ci < 16; ++ci) {
      float e = __expf(sc[r][lane + ci * 32] - mx);
      ev[ci] = e; sum += e;
    }
    for (int o = 16; o >= 1; o >>= 1) sum += __shfl_xor(sum, o, 32);
    float inv = 1.0f / sum;
#pragma unroll
    for (int ci = 0; ci < 16; ++ci)
      pb[r][lane + ci * 32] = f2bf(ev[ci] * inv);
  }
  __syncthreads();

  // ---- ctx = P @ V  (16x512 @ 512x64) ----
  v8f ct[4] = {};
  for (int kt = 0; kt < 512; kt += 32) {
    Frag ap;
    const unsigned short* pr = &pb[m16][kt];
    ap.q[0] = *(const U4*)(pr + half * 16);
    ap.q[1] = *(const U4*)(pr + half * 16 + 8);
#pragma unroll
    for (int dj = 0; dj < 4; ++dj) {
      Frag bv;
      const unsigned short* vp =
          vbf + ((size_t)(bb * 512 + kt + half * 16)) * 512 + h * 64 + dj * 16 + m16;
#pragma unroll
      for (int e2 = 0; e2 < 8; ++e2) {
        unsigned int lo = vp[(2 * e2) * 512];
        unsigned int hi = vp[(2 * e2 + 1) * 512];
        bv.u[e2] = lo | (hi << 16);
      }
      ct[dj] = wmma_bf16(ap, bv, ct[dj]);
    }
  }
#pragma unroll
  for (int dj = 0; dj < 4; ++dj) {
    int col = h * 64 + dj * 16 + m16;
#pragma unroll
    for (int r = 0; r < 8; ++r) {
      int row = rb + half * 8 + r;
      ctxbf[((size_t)(bb * 512 + row)) * 512 + col] = f2bf(ct[dj][r]);
    }
  }
}

// ---------------------------------------------------------------------------
// Local edge attention (heads 4..7). Block = 128 threads = 4 waves, one wave
// per head; one block per (b,i). 8 edges j=(i+1..i+8)%N; ef-weighted dots,
// softmax over 8, weighted sum of V. Writes ctx cols [256,512).
// ---------------------------------------------------------------------------
__global__ __launch_bounds__(128)
void local_attn(const unsigned short* __restrict__ qbf,
                const unsigned short* __restrict__ kbf,
                const unsigned short* __restrict__ vbf,
                const unsigned short* __restrict__ efb,
                unsigned short* __restrict__ ctxbf) {
  const int p    = blockIdx.x;            // 0..16383 = b*512+i
  const int bb   = p >> 9;
  const int i    = p & 511;
  const int w    = threadIdx.x >> 5;      // 0..3
  const int lane = threadIdx.x & 31;
  const int h    = 4 + w;
  const int t    = lane >> 2;             // edge 0..7
  const int s4   = lane & 3;              // 16-dim segment

  const int j    = (i + 1 + t) & 511;
  const size_t e = (size_t)p * 8 + t;

  const unsigned short* qp = qbf + ((size_t)p) * 512 + h * 64 + s4 * 16;
  const unsigned short* kp = kbf + ((size_t)(bb * 512 + j)) * 512 + h * 64 + s4 * 16;
  const unsigned short* ep = efb + e * 256 + w * 64 + s4 * 16;

  float acc = 0.0f;
#pragma unroll
  for (int d = 0; d < 16; ++d)
    acc += bf2f(qp[d]) * bf2f(kp[d]) * bf2f(ep[d]);
  acc *= 0.125f;                                   // 1/sqrt(DH) on q
  acc += __shfl_down(acc, 1, 4);
  acc += __shfl_down(acc, 2, 4);                   // lane s4==0 holds sl[t]

  float sl[8];
#pragma unroll
  for (int tt = 0; tt < 8; ++tt) sl[tt] = __shfl(acc, tt * 4, 32);
  float mx = sl[0];
#pragma unroll
  for (int tt = 1; tt < 8; ++tt) mx = fmaxf(mx, sl[tt]);
  float al[8], sum = 0.0f;
#pragma unroll
  for (int tt = 0; tt < 8; ++tt) { al[tt] = __expf(sl[tt] - mx); sum += al[tt]; }
  float inv = 1.0f / sum;

  float c0 = 0.0f, c1 = 0.0f;
#pragma unroll
  for (int tt = 0; tt < 8; ++tt) {
    int jj = (i + 1 + tt) & 511;
    const unsigned short* vp =
        vbf + ((size_t)(bb * 512 + jj)) * 512 + h * 64 + lane * 2;
    float a = al[tt] * inv;
    c0 += a * bf2f(vp[0]);
    c1 += a * bf2f(vp[1]);
  }
  unsigned short* cp = ctxbf + ((size_t)p) * 512 + h * 64 + lane * 2;
  cp[0] = f2bf(c0);
  cp[1] = f2bf(c1);
}

// ---------------------------------------------------------------------------
// Host-side orchestration
// ---------------------------------------------------------------------------
extern "C" void kernel_launch(void* const* d_in, const int* in_sizes, int n_in,
                              void* d_out, int out_size, void* d_ws, size_t ws_size,
                              hipStream_t stream) {
  (void)in_sizes; (void)n_in; (void)out_size; (void)ws_size;

  const float* key_x   = (const float*)d_in[0];
  const float* value_x = (const float*)d_in[1];
  const float* query_x = (const float*)d_in[2];
  const unsigned char* maskp = (const unsigned char*)d_in[3];
  const float* edge_feature  = (const float*)d_in[4];
  const int*   pairs   = (const int*)d_in[5];
  const float* Wq = (const float*)d_in[6];  const float* bq = (const float*)d_in[7];
  const float* Wk = (const float*)d_in[8];  const float* bk = (const float*)d_in[9];
  const float* Wv = (const float*)d_in[10]; const float* bv = (const float*)d_in[11];
  const float* Wo = (const float*)d_in[12]; const float* bo = (const float*)d_in[13];
  const float* ep_w1 = (const float*)d_in[14]; const float* ep_b1 = (const float*)d_in[15];
  const float* ep_w2 = (const float*)d_in[16]; const float* ep_b2 = (const float*)d_in[17];
  const float* eu_w1 = (const float*)d_in[18]; const float* eu_b1 = (const float*)d_in[19];
  const float* eu_w2 = (const float*)d_in[20]; const float* eu_b2 = (const float*)d_in[21];

  float* out_f = (float*)d_out;                         // [32,512,512]
  float* top_f = out_f + (size_t)32 * 512 * 512;        // [32,512,512]
  float* eu_f  = top_f + (size_t)32 * 512 * 512;        // [131072,512]

  // --- workspace carve-up (bf16 buffers, 256B aligned) ---
  char* ws = (char*)d_ws;
  size_t off = 0;
  auto take = [&](size_t bytes) -> void* {
    void* pp = ws + off;
    off += (bytes + 255) & ~(size_t)255;
    return pp;
  };
  const int Ee = 131072;
  unsigned short* wq_t   = (unsigned short*)take(512 * 512 * 2);
  unsigned short* wk_t   = (unsigned short*)take(512 * 512 * 2);
  unsigned short* wv_t   = (unsigned short*)take(512 * 512 * 2);
  unsigned short* wo_t   = (unsigned short*)take(512 * 512 * 2);
  unsigned short* epw1_t = (unsigned short*)take(512 * 512 * 2);
  unsigned short* epw2_t = (unsigned short*)take(512 * 256 * 2);
  unsigned short* euw1_t = (unsigned short*)take(1024 * 512 * 2);
  unsigned short* euw2_t = (unsigned short*)take(512 * 512 * 2);
  unsigned short* qbf    = (unsigned short*)take((size_t)16384 * 512 * 2);
  unsigned short* kbf    = (unsigned short*)take((size_t)16384 * 512 * 2);
  unsigned short* vbf    = (unsigned short*)take((size_t)16384 * 512 * 2);
  unsigned short* hid    = (unsigned short*)take((size_t)Ee * 512 * 2);  // reused
  unsigned short* efb    = (unsigned short*)take((size_t)Ee * 256 * 2);
  unsigned short* ctxb   = (unsigned short*)take((size_t)16384 * 512 * 2);
  unsigned short* outb   = (unsigned short*)take((size_t)16384 * 512 * 2);

  // --- weight conversion + transpose: W[K,N] -> WT[N,K] bf16 ---
  cvtT_f32_bf16<<<256, 256, 0, stream>>>(Wq,    wq_t,   512, 512);
  cvtT_f32_bf16<<<256, 256, 0, stream>>>(Wk,    wk_t,   512, 512);
  cvtT_f32_bf16<<<256, 256, 0, stream>>>(Wv,    wv_t,   512, 512);
  cvtT_f32_bf16<<<256, 256, 0, stream>>>(Wo,    wo_t,   512, 512);
  cvtT_f32_bf16<<<256, 256, 0, stream>>>(ep_w1, epw1_t, 512, 512);
  cvtT_f32_bf16<<<256, 256, 0, stream>>>(ep_w2, epw2_t, 512, 256);
  cvtT_f32_bf16<<<256, 256, 0, stream>>>(eu_w1, euw1_t, 1024, 512);
  cvtT_f32_bf16<<<256, 256, 0, stream>>>(eu_w2, euw2_t, 512, 512);

  // --- Q/K/V projections (fp32 A converted during staging) ---
  gemm_wmma_bf16<<<dim3(8, 128), 256, 0, stream>>>(
      query_x, nullptr, nullptr, nullptr, 0, wq_t, bq, nullptr, qbf,
      16384, 512, 512, 0);
  gemm_wmma_bf16<<<dim3(8, 128), 256, 0, stream>>>(
      key_x, nullptr, nullptr, nullptr, 0, wk_t, bk, nullptr, kbf,
      16384, 512, 512, 0);
  gemm_wmma_bf16<<<dim3(8, 128), 256, 0, stream>>>(
      value_x, nullptr, nullptr, nullptr, 0, wv_t, bv, nullptr, vbf,
      16384, 512, 512, 0);

  // --- edge feature MLP: hid = ssp(E @ ep_w1 + b);  efb = hid @ ep_w2 + b ---
  gemm_wmma_bf16<<<dim3(8, 1024), 256, 0, stream>>>(
      edge_feature, nullptr, nullptr, nullptr, 0, epw1_t, ep_b1, nullptr, hid,
      Ee, 512, 512, 1);
  gemm_wmma_bf16<<<dim3(4, 1024), 256, 0, stream>>>(
      nullptr, hid, nullptr, nullptr, 0, epw2_t, ep_b2, nullptr, efb,
      Ee, 256, 512, 0);

  // --- attention: global heads 0..3 (WMMA), local heads 4..7 ---
  global_attn<<<dim3(32, 4, 32), 32, 0, stream>>>(qbf, kbf, vbf, maskp, ctxb, top_f);
  local_attn<<<dim3(16384), 128, 0, stream>>>(qbf, kbf, vbf, efb, ctxb);

  // --- output projection: out = ctx @ Wo + bo (f32 to d_out, bf16 for nf) ---
  gemm_wmma_bf16<<<dim3(8, 128), 256, 0, stream>>>(
      nullptr, ctxb, nullptr, nullptr, 0, wo_t, bo, out_f, outb,
      16384, 512, 512, 0);

  // --- edge update MLP: gather-concat nf rows (async), then second layer ---
  gemm_wmma_bf16<<<dim3(8, 1024), 256, 0, stream>>>(
      nullptr, nullptr, outb, pairs, Ee, euw1_t, eu_b1, nullptr, hid,
      Ee, 512, 1024, 1);
  gemm_wmma_bf16<<<dim3(8, 1024), 256, 0, stream>>>(
      nullptr, hid, nullptr, nullptr, 0, euw2_t, eu_b2, eu_f, nullptr,
      Ee, 512, 512, 0);
}
